// BetaStateVAE_MLP_39075612459504
// MI455X (gfx1250) — matrix operand order repaired and checked by reading, not compiled, not run on hardware
//
#include <hip/hip_runtime.h>

typedef __attribute__((ext_vector_type(16))) __bf16       v16bf;
typedef __attribute__((ext_vector_type(8)))  float        v8f;
typedef __attribute__((ext_vector_type(4)))  unsigned int v4u;

union FragU { v16bf v; v4u q[2]; };

#define B_N   8192
#define HID   256
#define ZD    16
#define NG    16
#define TS    16      // samples per generator block
#define JS    264     // padded J row stride (bf16 elems) -> 528B, banks spread
#define HS    264     // padded h row stride (bf16 elems)
#define ENC_S 64      // samples per encoder block
#define EHS   260     // encoder h stride (f32)

static constexpr size_t XR_OFF = (size_t)B_N * 256;        // after jac (B,16,16)
static constexpr size_t MU_OFF = XR_OFF + (size_t)B_N * 16;
static constexpr size_t LV_OFF = MU_OFF + (size_t)B_N * 16;
static constexpr size_t Z_OFF  = LV_OFF + (size_t)B_N * 16;

// LDS layout for gen_kernel (bytes): Ja 135168 | Jb 135168 | hb 8448 | mask 16384
#define GEN_SMEM (2 * 256 * JS * 2 + TS * HS * 2 + TS * HID * 4)   // 295168
#define ENC_SMEM (2 * ENC_S * EHS * 4)                              // 133120

__device__ __forceinline__ v8f wmma_bf16(FragU a, FragU b, v8f c) {
  return __builtin_amdgcn_wmma_f32_16x16x32_bf16(false, a.v, false, b.v,
                                                 (short)0, c, false, false);
}

// ---------------- weight conversion: fp32 row-major -> bf16 col-major ----------
__global__ void convert_kernel(const float* __restrict__ gw0, const float* __restrict__ gw1,
                               const float* __restrict__ gw2, const float* __restrict__ gw3,
                               __bf16* __restrict__ WT1, __bf16* __restrict__ WT2,
                               __bf16* __restrict__ WT3, float* __restrict__ w4f) {
  const int i = blockIdx.x * 256 + threadIdx.x;
  if (i < NG * HID * HID) {                 // WT[g][n][k] = W[g][k][n]
    const int g = i >> 16, n = (i >> 8) & 255, k = i & 255;
    const size_t src = ((size_t)g * HID + k) * HID + n;
    WT2[i] = (__bf16)gw1[src];
    WT3[i] = (__bf16)gw2[src];
  }
  if (i < NG * HID * ZD) {                  // WT1[g][n][k16]
    const int g = i >> 12, n = (i >> 4) & 255, k = i & 15;
    WT1[i] = (__bf16)gw0[((size_t)g * ZD + k) * HID + n];
  }
  if (i < NG * HID) w4f[i] = gw3[i];        // (16,256,1) -> f32 copy
}

// ---------------- encoder: fp32 VALU (1% of FLOPs, keeps mu/logvar/z exact) ----
template <int IN, int OUT, bool ACT>
__device__ void enc_layer(const float* __restrict__ w, const float* __restrict__ b,
                          const float* __restrict__ hin, float* __restrict__ hout, int tid) {
  constexpr int CNT = OUT / 4;
  const int sid = tid >> 2;
  const int ob  = (tid & 3) * CNT;
  float acc[CNT];
#pragma unroll
  for (int j = 0; j < CNT; ++j) acc[j] = 0.f;
  for (int k = 0; k < IN; ++k) {
    const float hk = hin[sid * EHS + k];
    const float* wr = w + (size_t)k * OUT + ob;
#pragma unroll
    for (int j = 0; j < CNT; ++j) acc[j] += hk * wr[j];
  }
#pragma unroll
  for (int j = 0; j < CNT; ++j) {
    float v = acc[j] + b[ob + j];
    if (ACT) v = (v >= 0.f) ? v : 0.2f * v;
    hout[sid * EHS + ob + j] = v;
  }
}

__global__ void __launch_bounds__(256)
enc_kernel(const float* __restrict__ x, const float* __restrict__ eps,
           const float* __restrict__ w0, const float* __restrict__ b0,
           const float* __restrict__ w1, const float* __restrict__ b1,
           const float* __restrict__ w2, const float* __restrict__ b2,
           const float* __restrict__ w3, const float* __restrict__ b3,
           const float* __restrict__ w4, const float* __restrict__ b4,
           float* __restrict__ out, __bf16* __restrict__ zbf) {
  extern __shared__ char smem[];
  float* hA = (float*)smem;
  float* hB = hA + ENC_S * EHS;
  const int tid = threadIdx.x;
  const int gs0 = blockIdx.x * ENC_S;

  for (int i = tid; i < ENC_S * ZD; i += 256) {
    const int s = i >> 4, k = i & 15;
    hA[s * EHS + k] = x[(size_t)(gs0 + s) * ZD + k];
  }
  __syncthreads();
  enc_layer<16, 256, true>(w0, b0, hA, hB, tid);  __syncthreads();
  enc_layer<256, 256, true>(w1, b1, hB, hA, tid); __syncthreads();
  enc_layer<256, 256, true>(w2, b2, hA, hB, tid); __syncthreads();
  enc_layer<256, 256, true>(w3, b3, hB, hA, tid); __syncthreads();
  enc_layer<256, 32, false>(w4, b4, hA, hB, tid); __syncthreads();

  for (int i = tid; i < ENC_S * ZD; i += 256) {
    const int s = i >> 4, d = i & 15;
    const float mu = hB[s * EHS + d];
    const float lv = hB[s * EHS + 16 + d];
    const size_t gi = (size_t)(gs0 + s) * ZD + d;
    const float zz = mu + __expf(0.5f * lv) * eps[gi];
    out[MU_OFF + gi] = mu;
    out[LV_OFF + gi] = lv;
    out[Z_OFF + gi]  = zz;
    zbf[gi] = (__bf16)zz;
  }
}

// ---------------- generator forward layer (M=16) via WMMA ----------------------
__device__ void fwd_layer(const __bf16* __restrict__ WT, const float* __restrict__ bias,
                          __bf16* hb, float* maskLDS, int lane, int wave) {
  const int rr  = lane & 15;
  const int kbA = (lane < 16) ? 0 : 16;   // A frag: 16 consecutive K at gl*16
  const int kbB = (lane < 16) ? 0 : 16;   // B frag: 16 consecutive K at gl*16
  FragU A[8];
#pragma unroll
  for (int kk = 0; kk < 8; ++kk) {
    const __bf16* p = hb + rr * HS + kk * 32 + kbA;
    A[kk].q[0] = *(const v4u*)(p);
    A[kk].q[1] = *(const v4u*)(p + 8);
  }
  __syncthreads();   // everyone read h_prev before overwriting
#pragma unroll
  for (int t2 = 0; t2 < 2; ++t2) {
    const int nt = wave * 2 + t2;
    const int n  = nt * 16 + rr;
    v8f ae = {0.f, 0.f, 0.f, 0.f, 0.f, 0.f, 0.f, 0.f};
    v8f ao = {0.f, 0.f, 0.f, 0.f, 0.f, 0.f, 0.f, 0.f};
#pragma unroll
    for (int kk = 0; kk < 8; kk += 2) {     // 2 independent WMMA chains
      FragU B0, B1;
      const __bf16* p0 = WT + (size_t)n * HID + kk * 32 + kbB;
      B0.q[0] = *(const v4u*)(p0);
      B0.q[1] = *(const v4u*)(p0 + 8);
      const __bf16* p1 = p0 + 32;
      B1.q[0] = *(const v4u*)(p1);
      B1.q[1] = *(const v4u*)(p1 + 8);
      ae = wmma_bf16(A[kk], B0, ae);
      ao = wmma_bf16(A[kk + 1], B1, ao);
    }
    const v8f acc = ae + ao;
    const float bv = bias[n];
#pragma unroll
    for (int v = 0; v < 8; ++v) {
      const int s = v + ((lane < 16) ? 0 : 8);
      const float pre = acc[v] + bv;
      const float mk  = (pre >= 0.f) ? 1.f : 0.2f;
      maskLDS[s * HID + n] = mk;
      hb[s * HS + n] = (__bf16)(pre * mk);
    }
  }
}

// ------------- Jacobian layer: Jout = (Jin @ W) * mask, 256x256x256 bf16 -------
__device__ void jac_layer(const __bf16* __restrict__ Jin, __bf16* __restrict__ Jout,
                          const __bf16* __restrict__ WT, const float* __restrict__ maskLDS,
                          int lane, int wave) {
  const int rr  = lane & 15;
  const int kbA = (lane < 16) ? 0 : 16;
  const int kbB = (lane < 16) ? 0 : 16;
  FragU A[2][8];                           // 2 M-tiles resident -> W streamed once
#pragma unroll
  for (int mt = 0; mt < 2; ++mt)
#pragma unroll
    for (int kk = 0; kk < 8; ++kk) {
      const __bf16* p = Jin + ((wave * 2 + mt) * 16 + rr) * JS + kk * 32 + kbA;
      A[mt][kk].q[0] = *(const v4u*)(p);
      A[mt][kk].q[1] = *(const v4u*)(p + 8);
    }
  for (int nt = 0; nt < 16; ++nt) {
    // 4 independent accumulator chains (even/odd K per M-tile) to hide WMMA
    // latency at 2 waves/SIMD occupancy.
    v8f a0e = {0.f, 0.f, 0.f, 0.f, 0.f, 0.f, 0.f, 0.f};
    v8f a0o = {0.f, 0.f, 0.f, 0.f, 0.f, 0.f, 0.f, 0.f};
    v8f a1e = {0.f, 0.f, 0.f, 0.f, 0.f, 0.f, 0.f, 0.f};
    v8f a1o = {0.f, 0.f, 0.f, 0.f, 0.f, 0.f, 0.f, 0.f};
#pragma unroll
    for (int kk = 0; kk < 8; kk += 2) {
      FragU B0, B1;
      const __bf16* p0 = WT + (size_t)(nt * 16 + rr) * HID + kk * 32 + kbB;
      B0.q[0] = *(const v4u*)(p0);
      B0.q[1] = *(const v4u*)(p0 + 8);
      const __bf16* p1 = p0 + 32;
      B1.q[0] = *(const v4u*)(p1);
      B1.q[1] = *(const v4u*)(p1 + 8);
      a0e = wmma_bf16(A[0][kk], B0, a0e);
      a1e = wmma_bf16(A[1][kk], B0, a1e);
      a0o = wmma_bf16(A[0][kk + 1], B1, a0o);
      a1o = wmma_bf16(A[1][kk + 1], B1, a1o);
    }
    const v8f acc0 = a0e + a0o;
    const v8f acc1 = a1e + a1o;
#pragma unroll
    for (int mt = 0; mt < 2; ++mt) {
      const int s = wave * 2 + mt;                       // M-tile == sample
      const float mv = maskLDS[s * HID + nt * 16 + rr];  // mask indep. of tangent
      const v8f a = mt ? acc1 : acc0;
#pragma unroll
      for (int v = 0; v < 8; ++v) {
        const int t = v + ((lane < 16) ? 0 : 8);
        Jout[(s * 16 + t) * JS + nt * 16 + rr] = (__bf16)(a[v] * mv);
      }
    }
  }
}

__global__ void __launch_bounds__(256)
gen_kernel(const __bf16* __restrict__ zbf,
           const __bf16* __restrict__ WT1, const __bf16* __restrict__ WT2,
           const __bf16* __restrict__ WT3, const float* __restrict__ w4f,
           const float* __restrict__ gb0, const float* __restrict__ gb1,
           const float* __restrict__ gb2, const float* __restrict__ gb3,
           float* __restrict__ out) {
  extern __shared__ char smem[];
  __bf16* Ja = (__bf16*)smem;
  __bf16* Jb = Ja + 256 * JS;
  __bf16* hb = Jb + 256 * JS;
  float* maskLDS = (float*)(hb + TS * HS);

  const int tid  = threadIdx.x;
  const int lane = tid & 31;
  const int wave = tid >> 5;
  const int g    = blockIdx.y;
  const int s0   = blockIdx.x * TS;

  const __bf16* WT1g = WT1 + (size_t)g * HID * ZD;
  const __bf16* WT2g = WT2 + (size_t)g * HID * HID;
  const __bf16* WT3g = WT3 + (size_t)g * HID * HID;

  // Warm the near caches with both 128KB weight slabs (global_prefetch_b8;
  // does not touch LOADcnt). 256 threads x 4 lines covers all 1024 lines.
#pragma unroll
  for (int pf = 0; pf < 4; ++pf) {
    __builtin_prefetch((const char*)WT2g + ((size_t)pf * 256 + tid) * 128, 0, 1);
    __builtin_prefetch((const char*)WT3g + ((size_t)pf * 256 + tid) * 128, 0, 1);
  }

  // ---- layer 1: pre = z @ W1 + b0 (K=16, zero-padded to 32) ----
  {
    const int rr  = lane & 15;
    const int kbA = (lane < 16) ? 0 : 8;
    const v4u z4 = {0u, 0u, 0u, 0u};
    FragU A;
    A.q[0] = (lane < 16) ? *(const v4u*)(zbf + (size_t)(s0 + rr) * ZD) : z4;
    A.q[1] = (lane < 16) ? *(const v4u*)(zbf + (size_t)(s0 + rr) * ZD + 8) : z4;  // lanes>=16: K=16..31 zero
#pragma unroll
    for (int t2 = 0; t2 < 2; ++t2) {
      const int nt = wave * 2 + t2;
      const int n  = nt * 16 + rr;
      FragU Bf;
      if (lane < 16) {                         // lanes>=16 cover K=16..31 -> zero
        const __bf16* p = WT1g + n * ZD;
        Bf.q[0] = *(const v4u*)(p);
        Bf.q[1] = *(const v4u*)(p + 8);
      } else {
        Bf.q[0] = z4; Bf.q[1] = z4;
      }
      v8f acc = {0.f, 0.f, 0.f, 0.f, 0.f, 0.f, 0.f, 0.f};
      acc = wmma_bf16(A, Bf, acc);
      const float bv = gb0[g * HID + n];
#pragma unroll
      for (int v = 0; v < 8; ++v) {
        const int s = v + ((lane < 16) ? 0 : 8);
        const float pre = acc[v] + bv;
        const float mk  = (pre >= 0.f) ? 1.f : 0.2f;
        maskLDS[s * HID + n] = mk;
        hb[s * HS + n] = (__bf16)(pre * mk);
      }
    }
  }
  __syncthreads();

  // ---- J1[s*16+t][n] = W1[t][n] * mask1[s][n] ----
  {
    const int n = tid;
    union { v4u q[2]; __bf16 e[16]; } wc;
    const __bf16* p = WT1g + n * ZD;
    wc.q[0] = *(const v4u*)(p);
    wc.q[1] = *(const v4u*)(p + 8);
#pragma unroll 4
    for (int s = 0; s < 16; ++s) {
      const float m = maskLDS[s * HID + n];
#pragma unroll
      for (int t = 0; t < 16; ++t)
        Ja[(s * 16 + t) * JS + n] = (__bf16)((float)wc.e[t] * m);
    }
  }
  __syncthreads();

  fwd_layer(WT2g, gb1 + g * HID, hb, maskLDS, lane, wave);
  __syncthreads();
  jac_layer(Ja, Jb, WT2g, maskLDS, lane, wave);
  __syncthreads();
  fwd_layer(WT3g, gb2 + g * HID, hb, maskLDS, lane, wave);
  __syncthreads();
  jac_layer(Jb, Ja, WT3g, maskLDS, lane, wave);
  __syncthreads();

  // stage w4 column into LDS (mask buffer is dead now)
  maskLDS[tid] = w4f[g * HID + tid];
  __syncthreads();

  // jac_mat[b][g][t] = J3 row . w4
  {
    const int r = tid, s = r >> 4, t = r & 15;
    float acc = 0.f;
#pragma unroll 8
    for (int n = 0; n < HID; ++n) acc += (float)Ja[r * JS + n] * maskLDS[n];
    out[(size_t)(s0 + s) * 256 + (size_t)g * 16 + t] = acc;
  }
  // x_recon[b][g] = h3 . w4 + b4
  if (tid < TS) {
    float acc = 0.f;
#pragma unroll 8
    for (int n = 0; n < HID; ++n) acc += (float)hb[tid * HS + n] * maskLDS[n];
    out[XR_OFF + (size_t)(s0 + tid) * 16 + g] = acc + gb3[g];
  }
}

// ------------------------------- launch ---------------------------------------
extern "C" void kernel_launch(void* const* d_in, const int* in_sizes, int n_in,
                              void* d_out, int out_size, void* d_ws, size_t ws_size,
                              hipStream_t stream) {
  (void)in_sizes; (void)n_in; (void)out_size; (void)ws_size;
  const float* x   = (const float*)d_in[0];
  const float* eps = (const float*)d_in[1];
  const float* ew0 = (const float*)d_in[2];  const float* eb0 = (const float*)d_in[3];
  const float* ew1 = (const float*)d_in[4];  const float* eb1 = (const float*)d_in[5];
  const float* ew2 = (const float*)d_in[6];  const float* eb2 = (const float*)d_in[7];
  const float* ew3 = (const float*)d_in[8];  const float* eb3 = (const float*)d_in[9];
  const float* ew4 = (const float*)d_in[10]; const float* eb4 = (const float*)d_in[11];
  const float* gw0 = (const float*)d_in[12]; const float* gb0 = (const float*)d_in[13];
  const float* gw1 = (const float*)d_in[14]; const float* gb1 = (const float*)d_in[15];
  const float* gw2 = (const float*)d_in[16]; const float* gb2 = (const float*)d_in[17];
  const float* gw3 = (const float*)d_in[18]; const float* gb3 = (const float*)d_in[19];
  float* out = (float*)d_out;

  char* ws = (char*)d_ws;
  __bf16* zbf = (__bf16*)ws;                                        // 262144 B
  __bf16* WT1 = (__bf16*)(ws + 262144);                             // 131072 B
  __bf16* WT2 = (__bf16*)(ws + 262144 + 131072);                    // 2097152 B
  __bf16* WT3 = (__bf16*)(ws + 262144 + 131072 + 2097152);          // 2097152 B
  float*  w4f = (float*)(ws + 262144 + 131072 + 2 * 2097152);       // 16384 B

  convert_kernel<<<4096, 256, 0, stream>>>(gw0, gw1, gw2, gw3, WT1, WT2, WT3, w4f);

  (void)hipFuncSetAttribute(reinterpret_cast<const void*>(enc_kernel),
                            hipFuncAttributeMaxDynamicSharedMemorySize, ENC_SMEM);
  enc_kernel<<<B_N / ENC_S, 256, ENC_SMEM, stream>>>(
      x, eps, ew0, eb0, ew1, eb1, ew2, eb2, ew3, eb3, ew4, eb4, out, zbf);

  (void)hipFuncSetAttribute(reinterpret_cast<const void*>(gen_kernel),
                            hipFuncAttributeMaxDynamicSharedMemorySize, GEN_SMEM);
  gen_kernel<<<dim3(B_N / TS, NG), 256, GEN_SMEM, stream>>>(
      zbf, WT1, WT2, WT3, w4f, gb0, gb1, gb2, gb3, out);
}
